// VQ_VAE_61418032333357
// MI455X (gfx1250) — matrix operand order repaired and checked by reading, not compiled, not run on hardware
//
#include <hip/hip_runtime.h>
#include <stdint.h>
#include <math.h>

// ---------------------------------------------------------------------------
// Types for CDNA5 WMMA (wave32)
// ---------------------------------------------------------------------------
typedef __attribute__((ext_vector_type(16))) __bf16 v16bf;
typedef __attribute__((ext_vector_type(8)))  __bf16 v8bf;
typedef __attribute__((ext_vector_type(8)))  float  v8f;
typedef __attribute__((ext_vector_type(4)))  int    v4i;

static __device__ __forceinline__ uint16_t f2bf(float f) {
  union { float f; uint32_t u; } v; v.f = f;
  uint32_t r = v.u + 0x7FFFu + ((v.u >> 16) & 1u);   // round-to-nearest-even
  return (uint16_t)(r >> 16);
}

// Problem constants
#define BSZ   32768
#define INDIM 784
#define H0    512
#define H1    1024
#define KCODE 128
#define DDIM  128
#define NSLOT 8

#define LDP 40   // LDS row stride in halves (32 data + 8 pad -> kills 16-bank stride conflicts)

// Epilogue modes
#define EP_RELU_BF16 0
#define EP_F32       1
#define EP_TANH_F32  2

// ---------------------------------------------------------------------------
// CDNA5 async memory->LDS path (GLOBAL_LOAD_ASYNC_TO_LDS_B128, ASYNCcnt)
// Builtin signature (from compiler diagnostic): (v4i AS1*, v4i AS3*, Imm, Imm)
// ---------------------------------------------------------------------------
#if __has_builtin(__builtin_amdgcn_global_load_async_to_lds_b128)
#define HAS_ASYNC_LDS 1
#define AS1 __attribute__((address_space(1)))
#define AS3 __attribute__((address_space(3)))
static __device__ __forceinline__ void async_b128(const void* g, void* l) {
  __builtin_amdgcn_global_load_async_to_lds_b128((AS1 v4i*)g, (AS3 v4i*)l, 0, 0);
}
static __device__ __forceinline__ void wait_async0() {
#if __has_builtin(__builtin_amdgcn_s_wait_asynccnt)
  __builtin_amdgcn_s_wait_asynccnt(0);
#else
  asm volatile("s_wait_asynccnt 0x0" ::: "memory");
#endif
}
#else
#define HAS_ASYNC_LDS 0
#endif

// ---------------------------------------------------------------------------
// f32 -> bf16 conversion of x, padded K 784 -> 800
// ---------------------------------------------------------------------------
__global__ void cvt_x_kernel(const float* __restrict__ x, uint16_t* __restrict__ xb) {
  int i = blockIdx.x * 256 + threadIdx.x;         // over 32768*800
  int m = i / 800, k = i - m * 800;
  float v = (k < INDIM) ? x[(size_t)m * INDIM + k] : 0.0f;
  xb[i] = f2bf(v);
}

// Transpose-convert weight: src f32 [K][N] -> dst bf16 [Npad][Kpad], zero padded
__global__ void cvt_wT_kernel(const float* __restrict__ W, uint16_t* __restrict__ Wt,
                              int K, int N, int Kpad, int Npad) {
  int i = blockIdx.x * 256 + threadIdx.x;
  if (i >= Npad * Kpad) return;
  int n = i / Kpad, k = i - n * Kpad;
  float v = (n < N && k < K) ? W[(size_t)k * N + n] : 0.0f;
  Wt[i] = f2bf(v);
}

// e2[c] = sum_d E[d][c]^2
__global__ void e2_kernel(const float* __restrict__ E, float* __restrict__ e2) {
  int c = threadIdx.x;                    // 128 threads
  float s = 0.0f;
  for (int d = 0; d < DDIM; ++d) { float v = E[(size_t)d * KCODE + c]; s += v * v; }
  e2[c] = s;
}

// ---------------------------------------------------------------------------
// WMMA helpers: fragment assembly per CDNA5 wave32 layouts (CDNA5 ISA)
// ---------------------------------------------------------------------------
static __device__ __forceinline__ v16bf load_a_frag(const uint16_t* __restrict__ lds,
                                                    int row, int hh) {
  // A 16x32 bf16: lane m=L&15 holds K = {hh*8 .. hh*8+7} and {hh*8+16 .. hh*8+23}
  const uint16_t* p = lds + row * LDP + hh * 8;
  v8bf lo = *(const v8bf*)(const void*)(p);
  v8bf hi = *(const v8bf*)(const void*)(p + 16);
  return __builtin_shufflevector(lo, hi, 0,1,2,3,4,5,6,7,8,9,10,11,12,13,14,15);
}
static __device__ __forceinline__ v16bf load_b_frag(const uint16_t* __restrict__ lds,
                                                    int row, int hh) {
  // B 32x16 bf16 (stored K-major per column): lane n=L&15 holds K = {hh*8 .. hh*8+7} and {hh*8+16 .. hh*8+23}
  const uint16_t* p = lds + row * LDP + hh * 8;
  v8bf lo = *(const v8bf*)(const void*)(p);
  v8bf hi = *(const v8bf*)(const void*)(p + 16);
  return __builtin_shufflevector(lo, hi, 0,1,2,3,4,5,6,7,8,9,10,11,12,13,14,15);
}

// 8 WMMAs over a 16x128 strip with software-pipelined B-fragment loads:
// ds_load of fragment t+1 is in flight under v_wmma of fragment t.
static __device__ __forceinline__ void wmma_strip(v8f acc[8],
                                                  const uint16_t* __restrict__ la,
                                                  const uint16_t* __restrict__ lb,
                                                  int wave, int nn, int hh) {
  v16bf af = load_a_frag(la, wave * 16 + nn, hh);
  v16bf bc = load_b_frag(lb, nn, hh);
#pragma unroll
  for (int t = 0; t < 8; ++t) {
    v16bf bn;
    if (t < 7) bn = load_b_frag(lb, (t + 1) * 16 + nn, hh);
    acc[t] = __builtin_amdgcn_wmma_f32_16x16x32_bf16(
        false, af, false, bc, (short)0, acc[t], false, false);
    bc = bn;
  }
}

// ---------------------------------------------------------------------------
// Tiled bf16 WMMA GEMM: C[M,N] = act(A[M,Kpad] * Bt[N,Kpad]^T + bias)
// Block: 128x128 tile, 256 threads (8 waves), each wave: 16 rows x 128 cols.
// Double-buffered LDS with async global->LDS copies (ASYNCcnt) when available.
// ---------------------------------------------------------------------------
__launch_bounds__(256)
__global__ void gemm_bf16_kernel(const uint16_t* __restrict__ A,
                                 const uint16_t* __restrict__ Bt,
                                 const float* __restrict__ bias,
                                 float* __restrict__ outF,
                                 uint16_t* __restrict__ outB,
                                 int Kpad, int Nlog, int ldOut, int epmode) {
  __shared__ uint16_t la[2][128 * LDP];
  __shared__ uint16_t lb[2][128 * LDP];

  const int tid  = threadIdx.x;
  const int wave = tid >> 5;
  const int lane = tid & 31;
  const int hh   = lane >> 4;        // half-wave select
  const int nn   = lane & 15;
  const int m0   = blockIdx.y * 128;
  const int n0   = blockIdx.x * 128;

  v8f acc[8];
#pragma unroll
  for (int t = 0; t < 8; ++t)
#pragma unroll
    for (int e = 0; e < 8; ++e) acc[t][e] = 0.0f;

  // staging map: 4 threads per row, 8 halves (16B) each, 64 rows per pass (2 passes)
  const int srow = tid >> 2;
  const int scol = (tid & 3) * 8;
  const int ksteps = Kpad >> 5;

  const uint16_t* gA = A  + (size_t)(m0 + srow) * Kpad + scol;
  const uint16_t* gB = Bt + (size_t)(n0 + srow) * Kpad + scol;
  const size_t rjump = (size_t)64 * Kpad;

#if HAS_ASYNC_LDS
  // ---- async double-buffered pipeline ----
  {
    async_b128(gA,         &la[0][srow * LDP + scol]);
    async_b128(gA + rjump, &la[0][(srow + 64) * LDP + scol]);
    async_b128(gB,         &lb[0][srow * LDP + scol]);
    async_b128(gB + rjump, &lb[0][(srow + 64) * LDP + scol]);
  }
  for (int ks = 0; ks < ksteps; ++ks) {
    wait_async0();        // this wave's tile-ks writes to LDS complete
    __syncthreads();      // all waves' writes visible; prev readers of other buf done
    const int cur = ks & 1;
    if (ks + 1 < ksteps) {
      const int nb = cur ^ 1;
      const int kb = (ks + 1) << 5;
      async_b128(gA + kb,         &la[nb][srow * LDP + scol]);
      async_b128(gA + kb + rjump, &la[nb][(srow + 64) * LDP + scol]);
      async_b128(gB + kb,         &lb[nb][srow * LDP + scol]);
      async_b128(gB + kb + rjump, &lb[nb][(srow + 64) * LDP + scol]);
    }
    wmma_strip(acc, la[cur], lb[cur], wave, nn, hh);
  }
#else
  // ---- fallback: synchronous staging (single logical buffer) ----
  for (int ks = 0; ks < ksteps; ++ks) {
    const int kb = ks << 5;
    __syncthreads();
    {
      uint4 a0 = *(const uint4*)(const void*)(gA + kb);
      uint4 a1 = *(const uint4*)(const void*)(gA + kb + rjump);
      uint4 b0 = *(const uint4*)(const void*)(gB + kb);
      uint4 b1 = *(const uint4*)(const void*)(gB + kb + rjump);
      if (ks + 1 < ksteps) {
        __builtin_prefetch(gA + kb + 32, 0, 0);
        __builtin_prefetch(gB + kb + 32, 0, 0);
      }
      *(uint4*)(void*)(&la[0][srow * LDP + scol])        = a0;
      *(uint4*)(void*)(&la[0][(srow + 64) * LDP + scol]) = a1;
      *(uint4*)(void*)(&lb[0][srow * LDP + scol])        = b0;
      *(uint4*)(void*)(&lb[0][(srow + 64) * LDP + scol]) = b1;
    }
    __syncthreads();
    wmma_strip(acc, la[0], lb[0], wave, nn, hh);
  }
#endif

  // Epilogue. C layout: row = r + 8*hh (r = accumulator element), col = nn.
  const int rowbase = m0 + wave * 16 + hh * 8;
#pragma unroll
  for (int t = 0; t < 8; ++t) {
    const int col = n0 + t * 16 + nn;
    if (col >= Nlog) continue;
    const float bv = bias[col];
#pragma unroll
    for (int r = 0; r < 8; ++r) {
      const int row = rowbase + r;
      float v = acc[t][r] + bv;
      if (epmode == EP_RELU_BF16) {
        v = v > 0.0f ? v : 0.0f;
        outB[(size_t)row * ldOut + col] = f2bf(v);
      } else if (epmode == EP_F32) {
        outF[(size_t)row * ldOut + col] = v;
      } else { // EP_TANH_F32
        outF[(size_t)row * ldOut + col] = tanhf(v);
      }
    }
  }
}

// ---------------------------------------------------------------------------
// VQ: per slot-row (B*8 rows), cross = z_slot . E^T via WMMA, then
// argmin_k (e2[k] - 2*cross) with shuffle reduction. 128 rows x 128 codes/block.
// A tile needs a strided f32->bf16 gather (VALU), Et tile uses async copy.
// ---------------------------------------------------------------------------
__launch_bounds__(256)
__global__ void vq_kernel(const float* __restrict__ ze,      // [B, 1024] f32
                          const uint16_t* __restrict__ Et,   // [128][128] bf16 (E^T)
                          const float* __restrict__ e2,      // [128]
                          int* __restrict__ idx) {           // [B*8]
  __shared__ uint16_t la[128 * LDP];
  __shared__ uint16_t lb[128 * LDP];

  const int tid  = threadIdx.x;
  const int wave = tid >> 5;
  const int lane = tid & 31;
  const int hh   = lane >> 4;
  const int nn   = lane & 15;
  const int r0   = blockIdx.x * 128;        // slot-row base

  v8f acc[8];
#pragma unroll
  for (int t = 0; t < 8; ++t)
#pragma unroll
    for (int e = 0; e < 8; ++e) acc[t][e] = 0.0f;

  const int srow = tid >> 2;
  const int scol = (tid & 3) * 8;
  const int arow = tid >> 1;                // 2 threads per row
  const int aj0  = (tid & 1) * 16;          // 16 elements each

  // A-row geometry: row R = b*8+n ; A[R][d] = ze[b][d*8+n]
  const int gsr  = r0 + arow;
  const int b    = gsr >> 3;
  const int slot = gsr & 7;
  const float* zp = ze + (size_t)b * H1 + slot;

  for (int ks = 0; ks < 4; ++ks) {
    const int kb = ks << 5;
    __syncthreads();
    {
      const uint16_t* gb = Et + (size_t)srow * DDIM + kb + scol;
#if HAS_ASYNC_LDS
      async_b128(gb,                      &lb[srow * LDP + scol]);
      async_b128(gb + (size_t)64 * DDIM,  &lb[(srow + 64) * LDP + scol]);
#else
      uint4 b0 = *(const uint4*)(const void*)(gb);
      uint4 b1 = *(const uint4*)(const void*)(gb + (size_t)64 * DDIM);
      *(uint4*)(void*)(&lb[srow * LDP + scol])        = b0;
      *(uint4*)(void*)(&lb[(srow + 64) * LDP + scol]) = b1;
#endif
      // Stage A: strided f32 gather -> bf16 (VALU conversion, ds_store)
      __align__(16) uint16_t tmp[16];
#pragma unroll
      for (int j = 0; j < 16; ++j) tmp[j] = f2bf(zp[(size_t)(kb + aj0 + j) * NSLOT]);
      *(uint4*)(void*)(&la[arow * LDP + aj0])     = *(const uint4*)(const void*)(tmp);
      *(uint4*)(void*)(&la[arow * LDP + aj0 + 8]) = *(const uint4*)(const void*)(tmp + 8);
#if HAS_ASYNC_LDS
      wait_async0();
#endif
    }
    __syncthreads();
    wmma_strip(acc, la, lb, wave, nn, hh);
  }

  // dist = e2[code] - 2*cross (z^2 term constant per row). Local mins, then
  // reduce across the 16 lanes (same hh) that hold a given row's 128 codes.
  float bestv[8]; int bestc[8];
#pragma unroll
  for (int r = 0; r < 8; ++r) { bestv[r] = 3.4e38f; bestc[r] = 0; }
#pragma unroll
  for (int t = 0; t < 8; ++t) {
    const int code = t * 16 + nn;
    const float eb = e2[code];
#pragma unroll
    for (int r = 0; r < 8; ++r) {
      float d = eb - 2.0f * acc[t][r];
      if (d < bestv[r] || (d == bestv[r] && code < bestc[r])) { bestv[r] = d; bestc[r] = code; }
    }
  }
#pragma unroll
  for (int off = 8; off >= 1; off >>= 1) {
#pragma unroll
    for (int r = 0; r < 8; ++r) {
      float ov = __shfl_xor(bestv[r], off, 32);
      int   oc = __shfl_xor(bestc[r], off, 32);
      if (ov < bestv[r] || (ov == bestv[r] && oc < bestc[r])) { bestv[r] = ov; bestc[r] = oc; }
    }
  }
  if (nn == 0) {
#pragma unroll
    for (int r = 0; r < 8; ++r) {
      const int row = wave * 16 + hh * 8 + r;     // C-layout row
      idx[r0 + row] = bestc[r];
    }
  }
}

// ---------------------------------------------------------------------------
// Gather: emb[b,d,n] = E[d, idx[b,n]] (f32 out) and z_q bf16 for next GEMM
// ---------------------------------------------------------------------------
__global__ void gather_kernel(const float* __restrict__ E, const int* __restrict__ idx,
                              float* __restrict__ emb, uint16_t* __restrict__ zqb) {
  size_t g = (size_t)blockIdx.x * 256 + threadIdx.x;  // over B*1024
  int b = (int)(g >> 10);
  int c = (int)(g & 1023);
  int d = c >> 3, n = c & 7;
  int code = idx[b * NSLOT + n];
  float v = E[(size_t)d * KCODE + code];
  emb[g] = v;                 // emb flat [B][D][N] == [B][c]
  zqb[g] = f2bf(v);           // z_q flat [B][H1], col c = d*8+n
}

// ---------------------------------------------------------------------------
// Host launcher
// ---------------------------------------------------------------------------
extern "C" void kernel_launch(void* const* d_in, const int* in_sizes, int n_in,
                              void* d_out, int out_size, void* d_ws, size_t ws_size,
                              hipStream_t stream) {
  const float* x  = (const float*)d_in[0];
  const float* W1 = (const float*)d_in[1];
  const float* b1 = (const float*)d_in[2];
  const float* W2 = (const float*)d_in[3];
  const float* b2 = (const float*)d_in[4];
  const float* W3 = (const float*)d_in[5];
  const float* b3 = (const float*)d_in[6];
  const float* W4 = (const float*)d_in[7];
  const float* b4 = (const float*)d_in[8];
  const float* E  = (const float*)d_in[9];

  float* out   = (float*)d_out;
  float* recon = out;                                        // [B, 784]
  float* ze    = out + (size_t)BSZ * INDIM;                  // [B, 1024]
  float* emb   = ze  + (size_t)BSZ * H1;                     // [B, 128, 8]

  // Workspace layout (256B aligned)
  char* ws = (char*)d_ws;
  size_t off = 0;
  auto nxt = [&](size_t bytes) { size_t o = off; off += (bytes + 255) & ~(size_t)255; return o; };
  uint16_t* xb   = (uint16_t*)(ws + nxt((size_t)BSZ * 800 * 2));    // x bf16, K padded
  uint16_t* W1t  = (uint16_t*)(ws + nxt((size_t)512 * 800 * 2));    // [512][800]
  uint16_t* h1b  = (uint16_t*)(ws + nxt((size_t)BSZ * H0 * 2));
  uint16_t* W2t  = (uint16_t*)(ws + nxt((size_t)1024 * 512 * 2));   // [1024][512]
  uint16_t* W3t  = (uint16_t*)(ws + nxt((size_t)512 * 1024 * 2));   // [512][1024]
  uint16_t* W4t  = (uint16_t*)(ws + nxt((size_t)896 * 512 * 2));    // [896][512]
  uint16_t* Etb  = (uint16_t*)(ws + nxt((size_t)128 * 128 * 2));    // E^T bf16
  float*    e2b  = (float*)   (ws + nxt((size_t)128 * 4));
  int*      idxb = (int*)     (ws + nxt((size_t)BSZ * NSLOT * 4));
  uint16_t* zqb  = (uint16_t*)(ws + nxt((size_t)BSZ * H1 * 2));
  uint16_t* h3b  = (uint16_t*)(ws + nxt((size_t)BSZ * H0 * 2));
  (void)ws_size; (void)in_sizes; (void)n_in; (void)out_size;

  // 1) conversions (weights tiny; x one pass)
  cvt_x_kernel<<<(BSZ * 800) / 256, 256, 0, stream>>>(x, xb);
  cvt_wT_kernel<<<(512 * 800 + 255) / 256, 256, 0, stream>>>(W1, W1t, INDIM, H0, 800, 512);
  cvt_wT_kernel<<<(1024 * 512 + 255) / 256, 256, 0, stream>>>(W2, W2t, H0, H1, 512, 1024);
  cvt_wT_kernel<<<(512 * 1024 + 255) / 256, 256, 0, stream>>>(W3, W3t, H1, H0, 1024, 512);
  cvt_wT_kernel<<<(896 * 512 + 255) / 256, 256, 0, stream>>>(W4, W4t, H0, INDIM, 512, 896);
  cvt_wT_kernel<<<(128 * 128 + 255) / 256, 256, 0, stream>>>(E, Etb, DDIM, KCODE, 128, 128);
  e2_kernel<<<1, 128, 0, stream>>>(E, e2b);

  // 2) h1 = relu(x@W1 + b1)  [bf16 out]
  gemm_bf16_kernel<<<dim3(H0 / 128, BSZ / 128), 256, 0, stream>>>(
      xb, W1t, b1, nullptr, h1b, 800, H0, H0, EP_RELU_BF16);

  // 3) z_e = h1@W2 + b2  [f32 out -> d_out region]
  gemm_bf16_kernel<<<dim3(H1 / 128, BSZ / 128), 256, 0, stream>>>(
      h1b, W2t, b2, ze, nullptr, 512, H1, H1, EP_F32);

  // 4) nearest-code search (WMMA cross + shuffle argmin)
  vq_kernel<<<(BSZ * NSLOT) / 128, 256, 0, stream>>>(ze, Etb, e2b, idxb);

  // 5) gather codebook -> emb (f32) and z_q (bf16)
  gather_kernel<<<(BSZ * H1) / 256, 256, 0, stream>>>(E, idxb, emb, zqb);

  // 6) h3 = relu(z_q@W3 + b3)  [bf16 out]
  gemm_bf16_kernel<<<dim3(H0 / 128, BSZ / 128), 256, 0, stream>>>(
      zqb, W3t, b3, nullptr, h3b, 1024, H0, H0, EP_RELU_BF16);

  // 7) recon = tanh(h3@W4 + b4)  [f32 out, N=784 guarded, Npad=896]
  gemm_bf16_kernel<<<dim3(896 / 128, BSZ / 128), 256, 0, stream>>>(
      h3b, W4t, b4, recon, nullptr, 512, INDIM, INDIM, EP_TANH_F32);
}
